// Block_89532888252905
// MI455X (gfx1250) — compile-verified, matrix operand order checked by reading, not run on hardware
//
#include <hip/hip_runtime.h>
#include <hip/hip_bf16.h>

// ---------------------------------------------------------------------------
// Types for CDNA5 WMMA (wave32): v_wmma_f32_16x16x32_bf16
// ---------------------------------------------------------------------------
typedef __attribute__((ext_vector_type(16))) __bf16 v16bf;
typedef __attribute__((ext_vector_type(8)))  float  v8f;

union FragBF { unsigned int u[8]; v16bf v; };

__device__ __forceinline__ v8f wmma_bf16(v16bf a, v16bf b, v8f c) {
  // 8 args: (neg_a, A, neg_b, B, c_mod, C, reuse_a, reuse_b)
  return __builtin_amdgcn_wmma_f32_16x16x32_bf16(false, a, false, b,
                                                 (short)0, c, false, false);
}

__device__ __forceinline__ unsigned short f32_to_bf16(float f) {
  unsigned int u = __float_as_uint(f);
  unsigned int r = u + 0x7FFFu + ((u >> 16) & 1u);   // round-to-nearest-even
  return (unsigned short)(r >> 16);
}

// ---------------------------------------------------------------------------
// CDNA5 async global->LDS copy (GLOBAL_LOAD_ASYNC_TO_LDS_B128, ASYNCcnt).
// LDS destination is the per-lane 32-bit LDS byte offset in VDST.
// ---------------------------------------------------------------------------
__device__ __forceinline__ void async_load_b128(const void* gptr, void* lptr) {
  unsigned int ldsOff =
      (unsigned int)(size_t)(__attribute__((address_space(3))) char*)lptr;
  asm volatile("global_load_async_to_lds_b128 %0, %1, off"
               :: "v"(ldsOff), "v"(gptr) : "memory");
}

__device__ __forceinline__ void wait_async0() {
  asm volatile("s_wait_asynccnt 0x0" ::: "memory");
}

// Load a 16x32 bf16 fragment (A layout, or B layout with row==N) from LDS.
// lds points at uint-packed bf16 pairs; rowStrideU = row stride in uints.
// ISA 16-bit A layout (16x32): lane l: M=l%16; VGPR i holds K pair
//   k0 = (i<4?0:16) + (l>=16?8:0) + (i%4)*2  -> uint idx = k0/2.
__device__ __forceinline__ v16bf load_frag(const unsigned int* lds, int row0,
                                           int rowStrideU, int kBaseU) {
  const int lane = threadIdx.x & 31;
  const int m  = lane & 15;
  const int hi = lane >> 4;
  FragBF f;
#pragma unroll
  for (int i = 0; i < 8; ++i) {
    int kp = kBaseU + ((i < 4) ? 0 : 8) + hi * 4 + (i & 3);
    f.u[i] = lds[(row0 + m) * rowStrideU + kp];
  }
  return f.v;
}

// ---------------------------------------------------------------------------
// Problem constants
// ---------------------------------------------------------------------------
#define N_EMBD 768
#define N_HEAD 12
#define HS 64
#define SEQ 1024
#define BATCH 8
#define M_TOK (BATCH * SEQ)     // 8192 tokens
#define C3 (3 * N_EMBD)         // 2304
#define C4 (4 * N_EMBD)         // 3072

// ---------------------------------------------------------------------------
// Weight f32 [K x N] -> bf16 transposed [N x K], LDS-tiled (64x64 tiles).
// Grid: (N/64, K/64), 256 threads.
// ---------------------------------------------------------------------------
__global__ __launch_bounds__(256) void transpose_bf16_kernel(
    const float* __restrict__ src, unsigned short* __restrict__ dst,
    int K, int N) {
  __shared__ unsigned short tile[64 * 65];
  const int nBase = blockIdx.x * 64;
  const int kBase = blockIdx.y * 64;
  const int tid = threadIdx.x;
#pragma unroll
  for (int it = 0; it < 16; ++it) {
    int i = tid + it * 256;          // 0..4095
    int r = i >> 6, c = i & 63;      // r = k offset, c = n offset
    tile[r * 65 + c] = f32_to_bf16(src[(size_t)(kBase + r) * N + nBase + c]);
  }
  __syncthreads();
#pragma unroll
  for (int it = 0; it < 16; ++it) {
    int i = tid + it * 256;
    int r = i >> 6, c = i & 63;      // r = n offset, c = k offset
    dst[(size_t)(nBase + r) * K + kBase + c] = tile[c * 65 + r];
  }
}

// ---------------------------------------------------------------------------
// LayerNorm over C=768, one block per token, bf16 output
// ---------------------------------------------------------------------------
__global__ __launch_bounds__(256) void layernorm_bf16_kernel(
    const float* __restrict__ x, const float* __restrict__ g,
    const float* __restrict__ b, unsigned short* __restrict__ out) {
  __shared__ float red[256];
  const int tid = threadIdx.x;
  const float* xr = x + (size_t)blockIdx.x * N_EMBD;
  float v0 = xr[tid], v1 = xr[tid + 256], v2 = xr[tid + 512];

  red[tid] = v0 + v1 + v2;
  __syncthreads();
  for (int off = 128; off > 0; off >>= 1) {
    if (tid < off) red[tid] += red[tid + off];
    __syncthreads();
  }
  float mean = red[0] * (1.0f / N_EMBD);
  __syncthreads();

  float d0 = v0 - mean, d1 = v1 - mean, d2 = v2 - mean;
  red[tid] = d0 * d0 + d1 * d1 + d2 * d2;
  __syncthreads();
  for (int off = 128; off > 0; off >>= 1) {
    if (tid < off) red[tid] += red[tid + off];
    __syncthreads();
  }
  float rstd = rsqrtf(red[0] * (1.0f / N_EMBD) + 1e-5f);

  unsigned short* o = out + (size_t)blockIdx.x * N_EMBD;
  o[tid]       = f32_to_bf16(d0 * rstd * g[tid]       + b[tid]);
  o[tid + 256] = f32_to_bf16(d1 * rstd * g[tid + 256] + b[tid + 256]);
  o[tid + 512] = f32_to_bf16(d2 * rstd * g[tid + 512] + b[tid + 512]);
}

// ---------------------------------------------------------------------------
// GEMM: C[M,N] = A(bf16)[M,K] @ WT(bf16)[N,K]^T + bias, fused epilogue.
// 128 threads = 4 waves; tile 64x64; K-step 32.
// Async global->LDS double-buffered pipeline: one barrier per K-step; the
// async engine fills buf^1 while the waves run the WMMA chain on buf.
// epilogue: 0 = f32 store (+residual if non-null)
//           1 = bf16 store
//           2 = tanh-GELU -> bf16 store
// ---------------------------------------------------------------------------
__global__ __launch_bounds__(128) void gemm_bf16_kernel(
    const unsigned short* __restrict__ A, const unsigned short* __restrict__ WT,
    const float* __restrict__ bias, const float* __restrict__ residual,
    float* __restrict__ outF, unsigned short* __restrict__ outBf,
    int M, int N, int K, int epilogue) {
  __shared__ uint4 sA4[2][256];   // 64 rows x 16 uints (64x32 bf16), row-major
  __shared__ uint4 sB4[2][256];   // 64 n-rows x 16 uints (64x32 bf16), row-major

  const int tid  = threadIdx.x;
  const int wave = tid >> 5;
  const int lane = tid & 31;
  const int nBase = blockIdx.x * 64;
  const int mBase = blockIdx.y * 64;
  const int KU4 = K >> 3;                 // row stride in uint4
  const uint4* A4 = (const uint4*)A;
  const uint4* W4 = (const uint4*)WT;
  const int row0 = tid >> 2;              // 0..31
  const int quad = tid & 3;               // 0..3

  v8f acc[4];
#pragma unroll
  for (int t = 0; t < 4; ++t)
#pragma unroll
    for (int r = 0; r < 8; ++r) acc[t][r] = 0.0f;

  // Issue one 64x32 A tile + 64x32 B tile (4 b128 async loads per thread)
  auto issue_tile = [&](int k0, int buf) {
    const size_t kq = (size_t)(k0 >> 3);
    async_load_b128(&A4[(size_t)(mBase + row0)      * KU4 + kq + quad], &sA4[buf][tid]);
    async_load_b128(&A4[(size_t)(mBase + row0 + 32) * KU4 + kq + quad], &sA4[buf][tid + 128]);
    async_load_b128(&W4[(size_t)(nBase + row0)      * KU4 + kq + quad], &sB4[buf][tid]);
    async_load_b128(&W4[(size_t)(nBase + row0 + 32) * KU4 + kq + quad], &sB4[buf][tid + 128]);
  };

  issue_tile(0, 0);
  int buf = 0;
  for (int k0 = 0; k0 < K; k0 += 32) {
    wait_async0();        // this wave's fill of sX4[buf] complete
    __syncthreads();      // all waves' fills complete
    if (k0 + 32 < K) issue_tile(k0 + 32, buf ^ 1);   // overlap with WMMAs

    v16bf afrag = load_frag((const unsigned int*)sA4[buf], wave * 16, 16, 0);
    v16bf bfrag[4];
#pragma unroll
    for (int t = 0; t < 4; ++t)
      bfrag[t] = load_frag((const unsigned int*)sB4[buf], t * 16, 16, 0);
#pragma unroll
    for (int t = 0; t < 4; ++t)
      acc[t] = wmma_bf16(afrag, bfrag[t], acc[t]);
    buf ^= 1;
  }

  // Epilogue. C/D layout: VGPR r -> row mBase + wave*16 + r + (lane>=16 ? 8:0),
  // col nBase + t*16 + lane%16.
  const int mRow0 = mBase + wave * 16 + ((lane >> 4) ? 8 : 0);
  const int c0 = lane & 15;
#pragma unroll
  for (int t = 0; t < 4; ++t) {
    int n = nBase + t * 16 + c0;
    float bv = bias[n];
#pragma unroll
    for (int r = 0; r < 8; ++r) {
      size_t idx = (size_t)(mRow0 + r) * N + n;
      float v = acc[t][r] + bv;
      if (epilogue == 0) {
        if (residual) v += residual[idx];
        outF[idx] = v;
      } else if (epilogue == 1) {
        outBf[idx] = f32_to_bf16(v);
      } else {
        const float c = 0.7978845608028654f;  // sqrt(2/pi)
        float gel = 0.5f * v * (1.0f + tanhf(c * (v + 0.044715f * v * v * v)));
        outBf[idx] = f32_to_bf16(gel);
      }
    }
  }
}

// ---------------------------------------------------------------------------
// Causal flash attention. Grid: (T/64, N_HEAD, BATCH), 128 threads (4 waves).
// qkv: bf16 [8192 x 2304] (q|k|v). y: bf16 [8192 x 768].
// Q/K tiles use the async global->LDS path; V needs a transpose so it stays
// on the synchronous b32-load + b16-scatter path.
// ---------------------------------------------------------------------------
__global__ __launch_bounds__(128) void attention_kernel(
    const unsigned short* __restrict__ qkv, unsigned short* __restrict__ y) {
  __shared__ uint4 sQ4[512];                 // [q][feat], 64x64 bf16 row-major
  __shared__ uint4 sK4[512];                 // [key][feat], 64x64 bf16 row-major
  __shared__ unsigned short sVT[64 * 68];    // [feat][key], padded stride 68
  __shared__ float sS[64 * 64];              // scores
  __shared__ unsigned int  sP[64 * 32];      // [q][key-pair] bf16 probs
  __shared__ float sM[64], sL[64], sAlpha[64];

  const int tid = threadIdx.x, wave = tid >> 5, lane = tid & 31;
  const int qt = blockIdx.x;
  const int h  = blockIdx.y;
  const int bb = blockIdx.z;
  const size_t tokBase = (size_t)bb * SEQ;
  const int qBase = qt * 64;

  const uint4* qkv4 = (const uint4*)qkv;
  const unsigned int* qkv32 = (const unsigned int*)qkv;
  const int rowU  = C3 >> 1;                 // 1152 uints / token
  const int rowU4 = C3 >> 3;                 // 288 uint4 / token
  const int qOff4 = (h * HS) >> 3;           // h*8
  const int kOff4 = (N_EMBD + h * HS) >> 3;  // 96 + h*8
  const int vOffU = (2 * N_EMBD + h * HS) >> 1;  // 768 + h*32

  // Async-load Q tile (64 rows x 8 uint4/row)
#pragma unroll
  for (int it = 0; it < 4; ++it) {
    int i = tid + it * 128;        // 0..511
    int r = i >> 3, q = i & 7;
    async_load_b128(&qkv4[(tokBase + qBase + r) * rowU4 + qOff4 + q], &sQ4[i]);
  }
  if (tid < 64) { sM[tid] = -INFINITY; sL[tid] = 0.0f; }

  v8f o[4];
#pragma unroll
  for (int t = 0; t < 4; ++t)
#pragma unroll
    for (int r = 0; r < 8; ++r) o[t][r] = 0.0f;
  wait_async0();
  __syncthreads();

  for (int kt = 0; kt <= qt; ++kt) {
    const int kBase = kt * 64;
    // Async-load K tile row-major [key][feat]
#pragma unroll
    for (int it = 0; it < 4; ++it) {
      int i = tid + it * 128;
      int r = i >> 3, q = i & 7;
      async_load_b128(&qkv4[(tokBase + kBase + r) * rowU4 + kOff4 + q], &sK4[i]);
    }
    // V tile transposed [feat][key]: b32 global loads, b16 LDS scatter
#pragma unroll
    for (int it = 0; it < 16; ++it) {
      int i = tid + it * 128;        // 0..2047 uints
      int key = i >> 5, fu = i & 31; // feats 2*fu, 2*fu+1
      unsigned int v = qkv32[(tokBase + kBase + key) * rowU + vOffU + fu];
      sVT[(2 * fu)     * 68 + key] = (unsigned short)(v & 0xFFFFu);
      sVT[(2 * fu + 1) * 68 + key] = (unsigned short)(v >> 16);
    }
    wait_async0();
    __syncthreads();

    // S = Q @ K^T  (A = Q rows, B rows = keys; K-dim = 64 feats)
    v8f s[4];
#pragma unroll
    for (int t = 0; t < 4; ++t)
#pragma unroll
      for (int r = 0; r < 8; ++r) s[t][r] = 0.0f;
#pragma unroll
    for (int ks = 0; ks < 2; ++ks) {
      v16bf a = load_frag((const unsigned int*)sQ4, wave * 16, 32, ks * 16);
      v16bf bfrag[4];
#pragma unroll
      for (int t = 0; t < 4; ++t)
        bfrag[t] = load_frag((const unsigned int*)sK4, t * 16, 32, ks * 16);
#pragma unroll
      for (int t = 0; t < 4; ++t)
        s[t] = wmma_bf16(a, bfrag[t], s[t]);
    }
    {
      const int r0 = wave * 16 + ((lane >> 4) ? 8 : 0);
      const int c0 = lane & 15;
#pragma unroll
      for (int t = 0; t < 4; ++t)
#pragma unroll
        for (int r = 0; r < 8; ++r)
          sS[(r0 + r) * 64 + t * 16 + c0] = s[t][r] * 0.125f;  // 1/sqrt(64)
    }
    __syncthreads();

    // Online softmax, one thread per query row
    if (tid < 64) {
      const int kLim = (kt == qt) ? (tid + 1) : 64;  // causal: key <= query
      float mOld = sM[tid];
      float mNew = mOld;
      for (int k2 = 0; k2 < kLim; ++k2) mNew = fmaxf(mNew, sS[tid * 64 + k2]);
      float alpha = expf(mOld - mNew);
      float sum = 0.0f;
#pragma unroll
      for (int kp = 0; kp < 32; ++kp) {
        int ka = kp * 2, kb = kp * 2 + 1;
        float pa = (ka < kLim) ? expf(sS[tid * 64 + ka] - mNew) : 0.0f;
        float pb = (kb < kLim) ? expf(sS[tid * 64 + kb] - mNew) : 0.0f;
        sum += pa + pb;
        sP[tid * 32 + kp] =
            (unsigned int)f32_to_bf16(pa) | ((unsigned int)f32_to_bf16(pb) << 16);
      }
      sM[tid] = mNew;
      sL[tid] = sL[tid] * alpha + sum;
      sAlpha[tid] = alpha;
    }
    __syncthreads();

    // Rescale O, then O += P @ V
    {
      const int r0 = wave * 16 + ((lane >> 4) ? 8 : 0);
#pragma unroll
      for (int t = 0; t < 4; ++t)
#pragma unroll
        for (int r = 0; r < 8; ++r) o[t][r] *= sAlpha[r0 + r];

      const unsigned int* sVT32 = (const unsigned int*)sVT;
#pragma unroll
      for (int ks = 0; ks < 2; ++ks) {
        v16bf a = load_frag(sP, wave * 16, 32, ks * 16);
        v16bf bfrag[4];
#pragma unroll
        for (int t = 0; t < 4; ++t)
          bfrag[t] = load_frag(sVT32, t * 16, 34, ks * 16);
#pragma unroll
        for (int t = 0; t < 4; ++t)
          o[t] = wmma_bf16(a, bfrag[t], o[t]);
      }
    }
    __syncthreads();
  }

  // Normalize and store y (bf16)
  const int r0 = wave * 16 + ((lane >> 4) ? 8 : 0);
  const int c0 = lane & 15;
#pragma unroll
  for (int t = 0; t < 4; ++t)
#pragma unroll
    for (int r = 0; r < 8; ++r) {
      int q = r0 + r;
      float inv = 1.0f / sL[q];
      y[(tokBase + qBase + q) * N_EMBD + h * HS + t * 16 + c0] =
          f32_to_bf16(o[t][r] * inv);
    }
}

// ---------------------------------------------------------------------------
// Host-side launch
// ---------------------------------------------------------------------------
extern "C" void kernel_launch(void* const* d_in, const int* in_sizes, int n_in,
                              void* d_out, int out_size, void* d_ws, size_t ws_size,
                              hipStream_t stream) {
  const float* x        = (const float*)d_in[0];
  const float* ln1_g    = (const float*)d_in[1];
  const float* ln1_b    = (const float*)d_in[2];
  const float* w_attn   = (const float*)d_in[3];
  const float* b_attn   = (const float*)d_in[4];
  const float* w_proj   = (const float*)d_in[5];
  const float* b_proj   = (const float*)d_in[6];
  const float* ln2_g    = (const float*)d_in[7];
  const float* ln2_b    = (const float*)d_in[8];
  const float* w_fc     = (const float*)d_in[9];
  const float* b_fc     = (const float*)d_in[10];
  const float* w_fc_proj= (const float*)d_in[11];
  const float* b_fc_proj= (const float*)d_in[12];
  float* out = (float*)d_out;

  // Workspace carve-up (bf16 buffers)
  char* ws = (char*)d_ws;
  size_t off = 0;
  auto carve = [&](size_t elems) {
    unsigned short* p = (unsigned short*)(ws + off);
    off += ((elems * 2 + 255) & ~(size_t)255);
    return p;
  };
  unsigned short* xln    = carve((size_t)M_TOK * N_EMBD);  // ln1/ln2 out (reused)
  unsigned short* qkvB   = carve((size_t)M_TOK * C3);
  unsigned short* yB     = carve((size_t)M_TOK * N_EMBD);
  unsigned short* hB     = carve((size_t)M_TOK * C4);
  unsigned short* wAttnT = carve((size_t)N_EMBD * C3);   // [C3 x 768]
  unsigned short* wProjT = carve((size_t)N_EMBD * N_EMBD);
  unsigned short* wFcT   = carve((size_t)N_EMBD * C4);   // [C4 x 768]
  unsigned short* wFcPT  = carve((size_t)C4 * N_EMBD);   // [768 x C4]
  (void)ws_size; (void)in_sizes; (void)n_in; (void)out_size;

  // 1) weights f32 [K x N] -> bf16 transposed [N x K]
  auto convT = [&](const float* src, unsigned short* dst, int K, int N) {
    transpose_bf16_kernel<<<dim3(N / 64, K / 64), 256, 0, stream>>>(src, dst, K, N);
  };
  convT(w_attn,    wAttnT, N_EMBD, C3);
  convT(w_proj,    wProjT, N_EMBD, N_EMBD);
  convT(w_fc,      wFcT,   N_EMBD, C4);
  convT(w_fc_proj, wFcPT,  C4, N_EMBD);

  // 2) ln1(x) -> xln (bf16)
  layernorm_bf16_kernel<<<M_TOK, 256, 0, stream>>>(x, ln1_g, ln1_b, xln);

  // 3) qkv = xln @ w_attn + b_attn  (bf16 out)
  gemm_bf16_kernel<<<dim3(C3 / 64, M_TOK / 64), 128, 0, stream>>>(
      xln, wAttnT, b_attn, nullptr, nullptr, qkvB, M_TOK, C3, N_EMBD, 1);

  // 4) causal flash attention -> yB (bf16)
  attention_kernel<<<dim3(SEQ / 64, N_HEAD, BATCH), 128, 0, stream>>>(qkvB, yB);

  // 5) x2 = x + yB @ w_proj + b_proj  (f32 -> d_out)
  gemm_bf16_kernel<<<dim3(N_EMBD / 64, M_TOK / 64), 128, 0, stream>>>(
      yB, wProjT, b_proj, x, out, nullptr, M_TOK, N_EMBD, N_EMBD, 0);

  // 6) ln2(x2) -> xln (bf16, reuse)
  layernorm_bf16_kernel<<<M_TOK, 256, 0, stream>>>(out, ln2_g, ln2_b, xln);

  // 7) h = gelu(xln @ w_fc + b_fc)  (bf16 out)
  gemm_bf16_kernel<<<dim3(C4 / 64, M_TOK / 64), 128, 0, stream>>>(
      xln, wFcT, b_fc, nullptr, nullptr, hB, M_TOK, C4, N_EMBD, 2);

  // 8) out = x2 + h @ w_fc_proj + b_fc_proj  (f32, in-place residual on d_out)
  gemm_bf16_kernel<<<dim3(N_EMBD / 64, M_TOK / 64), 128, 0, stream>>>(
      hB, wFcPT, b_fc_proj, out, out, nullptr, M_TOK, N_EMBD, C4, 0);
}
